// SelfAttention_26379689132232
// MI455X (gfx1250) — matrix operand order repaired and checked by reading, not compiled, not run on hardware
//
#include <hip/hip_runtime.h>
#include <hip/hip_bf16.h>

typedef _Float16 h16 __attribute__((ext_vector_type(16)));
typedef _Float16 h8  __attribute__((ext_vector_type(8)));
typedef float    f8  __attribute__((ext_vector_type(8)));
typedef unsigned int u32x4 __attribute__((ext_vector_type(4)));
typedef int          i32x8 __attribute__((ext_vector_type(8)));
typedef int          i32x4 __attribute__((ext_vector_type(4)));

#define B_  4
#define C_  256
#define N_  4096
#define CK  32

static __device__ __forceinline__ f8 wmma_f16(h16 a, h16 b, f8 c) {
    // D = A(16x32 f16) * B(32x16 f16) + C(16x16 f32)
    return __builtin_amdgcn_wmma_f32_16x16x32_f16(false, a, false, b,
                                                  (short)0, c, false, false);
}

static __device__ __forceinline__ h16 cat8(h8 lo, h8 hi) {
    return __builtin_shufflevector(lo, hi, 0,1,2,3,4,5,6,7,8,9,10,11,12,13,14,15);
}

// TDM 2D tile load: Mem[ga + 2*(x + y*stride)] -> LDS[lds_off + 2*(x + y*tile_d0)]
// D# group0: count=1 | lds_addr | global_addr[56:0] | type=2
// D# group1: data_size=1(2B); tensor_dim0/1; tile_dim0/1; tensor_dim0_stride
static __device__ __forceinline__ void tdm_load_2d(
    unsigned lds_off, const void* gptr,
    unsigned tensor_d0, unsigned tensor_d1,
    unsigned tile_d0, unsigned tile_d1, unsigned stride0)
{
    const unsigned long long ga = (unsigned long long)(uintptr_t)gptr;
    u32x4 g0;
    g0.x = 1u;                                                  // count=1
    g0.y = lds_off;                                             // lds_addr
    g0.z = (unsigned)(ga & 0xFFFFFFFFu);                        // gaddr lo
    g0.w = (unsigned)((ga >> 32) & 0x01FFFFFFu) | 0x80000000u;  // gaddr hi | type=2
    i32x8 g1;
    g1[0] = (int)(1u << 16);                                    // data_size = 2 bytes
    g1[1] = (int)((tensor_d0 & 0xFFFFu) << 16);                 // tensor_dim0 lo16
    g1[2] = (int)((tensor_d0 >> 16) | ((tensor_d1 & 0xFFFFu) << 16));
    g1[3] = (int)((tensor_d1 >> 16) | (tile_d0 << 16));         // tile_dim0
    g1[4] = (int)(tile_d1);                                     // tile_dim1, tile_dim2=0
    g1[5] = (int)(stride0);                                     // dim0 stride lo32
    g1[6] = 0;
    g1[7] = 0;
    const i32x4 z4 = {0, 0, 0, 0};
    const i32x8 z8 = {0, 0, 0, 0, 0, 0, 0, 0};
    __builtin_amdgcn_tensor_load_to_lds(g0, g1, z4, z4, z8, 0);
}

// ---------------------------------------------------------------------------
// Projection kernel: Q = scale*(wq x + bq) -> Qh[B][N][32]   (f16)
//                    K =        wk x + bk  -> Kh[B][N][32]   (f16)
//                    V =        wv x + bv  -> Vh[B][C][N]    (f16)
// grid (32, 20, 4), block 256 (8 waves). Wave = one 16x16 output tile.
// ---------------------------------------------------------------------------
__global__ __launch_bounds__(256) void proj_kernel(
    const float* __restrict__ x,
    const float* __restrict__ wq, const float* __restrict__ bq,
    const float* __restrict__ wk, const float* __restrict__ bk,
    const float* __restrict__ wv, const float* __restrict__ bv,
    _Float16* __restrict__ Qh, _Float16* __restrict__ Kh,
    _Float16* __restrict__ Vh)
{
    const int lane = threadIdx.x & 31;
    const int wave = threadIdx.x >> 5;
    const int col  = lane & 15;
    const int hl   = lane >> 4;
    const int b    = blockIdx.z;
    const int ot   = blockIdx.y;                 // 0..19 output-channel tile
    const int n0   = (blockIdx.x * 8 + wave) * 16;

    const float* W; const float* bias; int brow; float scl; int dest;
    if (ot < 2)      { W = wq; bias = bq; brow = ot*16;       scl = 0.17677669529663687f; dest = 0; }
    else if (ot < 4) { W = wk; bias = bk; brow = (ot-2)*16;   scl = 1.0f;                 dest = 1; }
    else             { W = wv; bias = bv; brow = (ot-4)*16;   scl = 1.0f;                 dest = 2; }

    // A layout: lane row = lane&15; halves 0..7 -> K = 8*hl + j,
    //           halves 8..15 -> K = 16 + 8*hl + j.
    h16 aw[8];
    const float* wr0 = W + (size_t)(brow + col) * C_;
    #pragma unroll
    for (int cc = 0; cc < 8; ++cc) {
        const float* wr = wr0 + cc * 32;
        #pragma unroll
        for (int j = 0; j < 8; ++j) {
            aw[cc][j]     = (_Float16)(wr[8*hl + j]      * scl);
            aw[cc][j + 8] = (_Float16)(wr[16 + 8*hl + j] * scl);
        }
    }

    const float* xb = x + (size_t)b * C_ * N_;
    f8 acc = {};
    #pragma unroll
    for (int cc = 0; cc < 8; ++cc) {
        // B layout: lane col = lane&15; halves i<8 -> K = 8*hl + i, i>=8 -> K = 16 + 8*hl + (i-8) (c index)
        h16 bx;
        #pragma unroll
        for (int i = 0; i < 16; ++i)
            bx[i] = (_Float16)xb[(size_t)(cc*32 + 8*hl + (i & 7) + 16*(i >> 3)) * N_ + n0 + col];
        acc = wmma_f16(aw[cc], bx, acc);
    }

    // bias (C layout: VGPR v -> row brow + 8*hl + v)
    #pragma unroll
    for (int v = 0; v < 8; ++v)
        acc[v] += bias[brow + 8*hl + v] * scl;

    if (dest == 2) {
        #pragma unroll
        for (int v = 0; v < 8; ++v) {
            const int o = brow + 8*hl + v;
            Vh[((size_t)b * C_ + o) * N_ + n0 + col] = (_Float16)acc[v];
        }
    } else {
        h8 pk;
        #pragma unroll
        for (int v = 0; v < 8; ++v) pk[v] = (_Float16)acc[v];
        _Float16* dst = (dest == 0) ? Qh : Kh;
        *(h8*)(dst + ((size_t)b * N_ + n0 + col) * CK + brow + 8*hl) = pk;
    }
}

// ---------------------------------------------------------------------------
// Flash-attention kernel. grid (32, 4), block 256 (8 waves).
// Wave owns 16 query rows; WG shares TDM-staged K/V chunks (double-buffered).
// ---------------------------------------------------------------------------
__global__ __launch_bounds__(256) void attn_kernel(
    const float* __restrict__ x,
    const float* __restrict__ gamma,
    const _Float16* __restrict__ Qh,
    const _Float16* __restrict__ Kh,
    const _Float16* __restrict__ Vh,
    float* __restrict__ out)
{
    __shared__ __align__(16) _Float16 ldsV[2][C_][32];   // 2 x 16 KB, [d][m]
    __shared__ __align__(16) _Float16 ldsK[2][32][CK];   // 2 x  2 KB, [m][ck]
    __shared__ __align__(16) _Float16 ldsP[8][16][32];   // per-wave P tile

    const int lane = threadIdx.x & 31;
    const int wave = threadIdx.x >> 5;
    const int col  = lane & 15;
    const int hl   = lane >> 4;
    const int b    = blockIdx.y;
    const int n0   = (blockIdx.x * 8 + wave) * 16;

    const _Float16* Kb = Kh + (size_t)b * N_ * CK;
    const _Float16* Vb = Vh + (size_t)b * C_ * N_;

    // Q A-fragment (softmax scale folded in at projection time)
    const _Float16* qrow = Qh + ((size_t)b * N_ + n0 + col) * CK;
    const h16 aq = cat8(*(const h8*)(qrow + 8*hl),
                        *(const h8*)(qrow + 16 + 8*hl));

    float m_i[8], l_i[8];
    #pragma unroll
    for (int v = 0; v < 8; ++v) { m_i[v] = -1e30f; l_i[v] = 0.0f; }
    f8 acc[16];
    #pragma unroll
    for (int t = 0; t < 16; ++t) acc[t] = (f8){};
    const f8 zero = {};

    // Prologue: DMA chunk 0 into buffer 0 (wave0: V tile, wave1: K tile)
    if (wave == 0)
        tdm_load_2d((unsigned)(uintptr_t)&ldsV[0][0][0], Vb,
                    N_, C_, 32, C_, N_);
    else if (wave == 1)
        tdm_load_2d((unsigned)(uintptr_t)&ldsK[0][0][0], Kb,
                    N_ * CK, 1, 1024, 1, N_ * CK);

    for (int it = 0; it < N_ / 32; ++it) {
        const int m0  = it * 32;
        const int cur = it & 1;

        if (it + 1 < N_ / 32) {   // prefetch next chunk into the other buffer
            if (wave == 0)
                tdm_load_2d((unsigned)(uintptr_t)&ldsV[cur ^ 1][0][0],
                            Vb + (m0 + 32), N_, C_, 32, C_, N_);
            else if (wave == 1)
                tdm_load_2d((unsigned)(uintptr_t)&ldsK[cur ^ 1][0][0],
                            Kb + (size_t)(m0 + 32) * CK, N_ * CK, 1, 1024, 1, N_ * CK);
            if (wave < 2) __builtin_amdgcn_s_wait_tensorcnt(1);
        } else {
            if (wave < 2) __builtin_amdgcn_s_wait_tensorcnt(0);
        }
        __syncthreads();                           // buf[cur] visible to all waves

        // K B-fragments from LDS: key cols = col / 16+col, K dim = head dim 32
        const _Float16* k0p = &ldsK[cur][col][8*hl];
        const _Float16* k1p = &ldsK[cur][16 + col][8*hl];
        const h16 bk0 = cat8(*(const h8*)k0p, *(const h8*)(k0p + 16));
        const h16 bk1 = cat8(*(const h8*)k1p, *(const h8*)(k1p + 16));

        f8 s0 = wmma_f16(aq, bk0, zero);
        f8 s1 = wmma_f16(aq, bk1, zero);

        // Online softmax. C layout: VGPR v holds row v+8*hl; the row's 16
        // key-cols live across the 16-lane half-group -> shfl_xor 1,2,4,8.
        float al[8], p0[8], p1[8];
        #pragma unroll
        for (int v = 0; v < 8; ++v) {
            float tm = fmaxf(s0[v], s1[v]);
            tm = fmaxf(tm, __shfl_xor(tm, 1, 32));
            tm = fmaxf(tm, __shfl_xor(tm, 2, 32));
            tm = fmaxf(tm, __shfl_xor(tm, 4, 32));
            tm = fmaxf(tm, __shfl_xor(tm, 8, 32));
            const float mn = fmaxf(m_i[v], tm);
            al[v] = __expf(m_i[v] - mn);
            m_i[v] = mn;
            p0[v] = __expf(s0[v] - mn);
            p1[v] = __expf(s1[v] - mn);
            float rs = p0[v] + p1[v];
            rs += __shfl_xor(rs, 1, 32);
            rs += __shfl_xor(rs, 2, 32);
            rs += __shfl_xor(rs, 4, 32);
            rs += __shfl_xor(rs, 8, 32);
            l_i[v] = l_i[v] * al[v] + rs;
        }
        #pragma unroll
        for (int t = 0; t < 16; ++t)
            #pragma unroll
            for (int v = 0; v < 8; ++v)
                acc[t][v] *= al[v];

        // Transpose P (C layout -> A layout) through wave-private LDS.
        // Same-wave DS ops are processed in order; fence stops reordering.
        #pragma unroll
        for (int v = 0; v < 8; ++v) {
            ldsP[wave][8*hl + v][col]      = (_Float16)p0[v];
            ldsP[wave][8*hl + v][16 + col] = (_Float16)p1[v];
        }
        __builtin_amdgcn_fence(__ATOMIC_ACQ_REL, "wavefront");
        const _Float16* pr = &ldsP[wave][col][0];
        const h16 ap = cat8(*(const h8*)(pr + 8*hl),
                            *(const h8*)(pr + 16 + 8*hl));

        // acc[16 n x 16 d] += P(16 x 32 keys) * V^T(32 keys x 16 d)
        #pragma unroll
        for (int dt = 0; dt < 16; ++dt) {
            const _Float16* vp = &ldsV[cur][dt*16 + col][8*hl];
            const h16 bv = cat8(*(const h8*)vp, *(const h8*)(vp + 16));
            acc[dt] = wmma_f16(ap, bv, acc[dt]);
        }
        __syncthreads();   // all waves done with buf[cur] before it is re-filled
    }

    // Epilogue: out[b][d][n] = gamma * acc[n][d] / l[n] + x[b][d][n]
    const float g = gamma[0];
    float inv[8];
    #pragma unroll
    for (int v = 0; v < 8; ++v) inv[v] = g / l_i[v];

    #pragma unroll
    for (int dt = 0; dt < 16; ++dt) {
        const int d = dt*16 + col;
        const size_t base = ((size_t)b * C_ + d) * N_ + n0 + 8*hl;
        const float4 x0 = *(const float4*)(x + base);
        const float4 x1 = *(const float4*)(x + base + 4);
        float4 o0, o1;
        o0.x = acc[dt][0]*inv[0] + x0.x;  o0.y = acc[dt][1]*inv[1] + x0.y;
        o0.z = acc[dt][2]*inv[2] + x0.z;  o0.w = acc[dt][3]*inv[3] + x0.w;
        o1.x = acc[dt][4]*inv[4] + x1.x;  o1.y = acc[dt][5]*inv[5] + x1.y;
        o1.z = acc[dt][6]*inv[6] + x1.z;  o1.w = acc[dt][7]*inv[7] + x1.w;
        *(float4*)(out + base)     = o0;
        *(float4*)(out + base + 4) = o1;
    }
}

// ---------------------------------------------------------------------------
extern "C" void kernel_launch(void* const* d_in, const int* in_sizes, int n_in,
                              void* d_out, int out_size, void* d_ws, size_t ws_size,
                              hipStream_t stream) {
    const float* x     = (const float*)d_in[0];
    const float* wq    = (const float*)d_in[1];
    const float* bq    = (const float*)d_in[2];
    const float* wk    = (const float*)d_in[3];
    const float* bk    = (const float*)d_in[4];
    const float* wv    = (const float*)d_in[5];
    const float* bv    = (const float*)d_in[6];
    const float* gamma = (const float*)d_in[7];
    float* out = (float*)d_out;

    char* ws = (char*)d_ws;
    const size_t qk_bytes = (size_t)B_ * N_ * CK * sizeof(_Float16);   // 1 MB each
    _Float16* Qh = (_Float16*)ws;
    _Float16* Kh = (_Float16*)(ws + qk_bytes);
    _Float16* Vh = (_Float16*)(ws + 2 * qk_bytes);                     // 8 MB

    proj_kernel<<<dim3(N_/128, 20, B_), 256, 0, stream>>>(
        x, wq, bq, wk, bk, wv, bv, Qh, Kh, Vh);
    attn_kernel<<<dim3(N_/128, B_), 256, 0, stream>>>(
        x, gamma, Qh, Kh, Vh, out);
}
